// ELISSABETH_833223656507
// MI455X (gfx1250) — matrix operand order repaired and checked by reading, not compiled, not run on hardware
//
#include <hip/hip_runtime.h>
#include <cstdint>
#include <cstddef>

#define T_SEQ   2048
#define BATCH   2
#define DHID    1024
#define DHEAD   512
#define NLAYERS 4
#define VOUT    32000
#define MTOT    (BATCH * T_SEQ)      // 4096 rows (b,t)
#define NV      (NLAYERS * DHEAD)    // 2048 combined (l,h)

typedef __attribute__((ext_vector_type(16))) __bf16 bf16x16;
typedef __attribute__((ext_vector_type(8)))  float  f32x8;

// ---------------------------------------------------------------------------
// WMMA fragment loads from a precomputed per-lane base pointer (keeps the
// k-offset as an immediate in the load instruction).  Layout per ISA 7.12.2:
//   A (16x32): lanes 0-15: elems[0..7]=K 0..7, [8..15]=K 16..23
//              lanes 16-31: K 8..15 and 24..31.   (base pre-offset by 0/8)
//   B (32x16): same K order as A: lanes 0-15: K 0..7,16..23; lanes 16-31: K 8..15,24..31.
//              (base pre-offset by 0/8; B stored N-major, K contiguous.)
// ---------------------------------------------------------------------------
__device__ __forceinline__ bf16x16 load_a_at(const __bf16* p, int kt) {
  union { bf16x16 v; uint4 q[2]; } u;
  u.q[0] = *reinterpret_cast<const uint4*>(p + kt);        // K +0..7
  u.q[1] = *reinterpret_cast<const uint4*>(p + kt + 16);   // K +16..23
  return u.v;
}
__device__ __forceinline__ bf16x16 load_b_at(const __bf16* p, int kt) {
  union { bf16x16 v; uint4 q[2]; } u;
  u.q[0] = *reinterpret_cast<const uint4*>(p + kt);        // K +0..7
  u.q[1] = *reinterpret_cast<const uint4*>(p + kt + 16);   // K +16..23
  return u.v;
}

// ---------------------------------------------------------------------------
// K0: x = emb[tok] + pos (fp32 -> bf16 copy) and q,k scalars per token/layer.
// ---------------------------------------------------------------------------
__global__ void embed_qk_kernel(const int* __restrict__ tokens,
                                const float* __restrict__ emb,
                                const float* __restrict__ pos,
                                const float* __restrict__ WQ,
                                const float* __restrict__ WK,
                                __bf16* __restrict__ xbf,
                                float* __restrict__ qout,
                                float* __restrict__ kout) {
  __shared__ float red[256][8];
  int m   = blockIdx.x;                // 0..4095  (b*T + t)
  int t   = m & (T_SEQ - 1);
  int tid = threadIdx.x;
  int tok = tokens[m];
  const float* e = emb + (size_t)tok * DHID;
  const float* p = pos + (size_t)t * DHID;

  float acc[8];
#pragma unroll
  for (int j = 0; j < 8; ++j) acc[j] = 0.f;

  for (int d = tid; d < DHID; d += 256) {
    float v = e[d] + p[d];
    xbf[(size_t)m * DHID + d] = (__bf16)v;
#pragma unroll
    for (int l = 0; l < NLAYERS; ++l) {
      acc[l]     += v * WQ[l * DHID + d];
      acc[4 + l] += v * WK[l * DHID + d];
    }
  }
#pragma unroll
  for (int j = 0; j < 8; ++j) red[tid][j] = acc[j];
  __syncthreads();
  for (int s = 128; s > 0; s >>= 1) {
    if (tid < s) {
#pragma unroll
      for (int j = 0; j < 8; ++j) red[tid][j] += red[tid + s][j];
    }
    __syncthreads();
  }
  if (tid < 4)             qout[(size_t)m * 4 + tid]        = red[0][tid];
  if (tid >= 4 && tid < 8) kout[(size_t)m * 4 + (tid - 4)]  = red[0][tid];
}

// ---------------------------------------------------------------------------
// K1: generic fp32 -> bf16 transpose (dst[c][r] = src[r][c]), 32x32 LDS tiles.
// ---------------------------------------------------------------------------
__global__ void transpose_f32_bf16(const float* __restrict__ src,
                                   __bf16* __restrict__ dst,
                                   int rows, int cols,
                                   size_t src_batch, size_t dst_batch) {
  src += (size_t)blockIdx.z * src_batch;
  dst += (size_t)blockIdx.z * dst_batch;
  __shared__ float tile[32][33];
  int tx = threadIdx.x, ty = threadIdx.y;
  int c0 = blockIdx.x * 32;
  int r0 = blockIdx.y * 32;
  for (int i = ty; i < 32; i += 8)
    tile[i][tx] = src[(size_t)(r0 + i) * cols + (c0 + tx)];
  __syncthreads();
  for (int i = ty; i < 32; i += 8)
    dst[(size_t)(c0 + i) * rows + (r0 + tx)] = (__bf16)tile[tx][i];
}

// ---------------------------------------------------------------------------
// K2: V projection, 4x4 register blocking per wave (wave tile 64x64,
// block tile 128x256).  Output stored transposed (t-contiguous) via LDS.
// ---------------------------------------------------------------------------
__global__ __launch_bounds__(256, 1)
void gemm_xv_kernel(const __bf16* __restrict__ X,
                    const __bf16* __restrict__ WVt,
                    float* __restrict__ Vt) {
  __shared__ float tile[8][16][17];
  int lane = threadIdx.x & 31, wave = threadIdx.x >> 5;
  int mtile = blockIdx.x * 128 + (wave >> 2) * 64;
  int ntile = blockIdx.y * 256 + (wave & 3) * 64;
  int lm  = lane & 15;
  int hiA = (lane >= 16) ? 8 : 0;
  int hiB = (lane >= 16) ? 8 : 0;

  const __bf16* pa[4];
  const __bf16* pb[4];
#pragma unroll
  for (int i = 0; i < 4; ++i)
    pa[i] = X + (size_t)(mtile + i * 16 + lm) * DHID + hiA;
#pragma unroll
  for (int j = 0; j < 4; ++j)
    pb[j] = WVt + (size_t)(ntile + j * 16 + lm) * DHID + hiB;

  f32x8 c[4][4];
#pragma unroll
  for (int i = 0; i < 4; ++i)
#pragma unroll
    for (int j = 0; j < 4; ++j)
#pragma unroll
      for (int r = 0; r < 8; ++r) c[i][j][r] = 0.f;

#pragma unroll 1
  for (int kt = 0; kt < DHID; kt += 32) {
    bf16x16 a[4], b[4];
#pragma unroll
    for (int i = 0; i < 4; ++i) a[i] = load_a_at(pa[i], kt);
#pragma unroll
    for (int j = 0; j < 4; ++j) b[j] = load_b_at(pb[j], kt);
#pragma unroll
    for (int i = 0; i < 4; ++i)
#pragma unroll
      for (int j = 0; j < 4; ++j)
        c[i][j] = __builtin_amdgcn_wmma_f32_16x16x32_bf16(
            false, a[i], false, b[j], (short)0, c[i][j], false, false);
  }

  int mo = (lane >= 16) ? 8 : 0;
  int nl = lane >> 1;           // local n in tile
  int mh = (lane & 1) * 8;      // m half
#pragma unroll
  for (int i = 0; i < 4; ++i) {
#pragma unroll
    for (int j = 0; j < 4; ++j) {
      __syncthreads();
#pragma unroll
      for (int r = 0; r < 8; ++r) tile[wave][r + mo][lm] = c[i][j][r];
      __syncthreads();
      float* dst = Vt + (size_t)(ntile + j * 16 + nl) * MTOT + mtile + i * 16 + mh;
#pragma unroll
      for (int u = 0; u < 8; ++u) dst[u] = tile[wave][mh + u][nl];
    }
  }
}

// ---------------------------------------------------------------------------
// K3: LISS scan, one layer per launch.  One block per (b,h) channel.
// ---------------------------------------------------------------------------
__global__ void scan_layer_kernel(const float* __restrict__ Vt,
                                  const float* __restrict__ q,
                                  const float* __restrict__ k,
                                  float* __restrict__ R,
                                  __bf16* __restrict__ Rbf,
                                  int layer) {
  __shared__ float s[256];
  int bh = blockIdx.x;
  int b  = bh >> 9;          // /512
  int h  = bh & (DHEAD - 1);
  int tid = threadIdx.x;
  int tbase = tid * 8;
  const float* vrow = Vt + (size_t)(layer * DHEAD + h) * MTOT + (size_t)b * T_SEQ;

  float a[8];
#pragma unroll
  for (int i = 0; i < 8; ++i) {
    int t = tbase + i;
    int m = b * T_SEQ + t;
    float v = vrow[t];
    float g = (layer == 0) ? __expf(-k[(size_t)m * 4])
                           : __expf(q[(size_t)m * 4 + layer - 1] - k[(size_t)m * 4 + layer]);
    float prev = 1.f;
    if (layer > 0) prev = (t == 0) ? 0.f : R[(size_t)(m - 1) * DHEAD + h];
    a[i] = g * v * prev;
  }
  __syncthreads();                         // all R reads done before overwrite
#pragma unroll
  for (int i = 1; i < 8; ++i) a[i] += a[i - 1];
  s[tid] = a[7];
  __syncthreads();
  for (int off = 1; off < 256; off <<= 1) {
    float t = (tid >= off) ? s[tid - off] : 0.f;
    __syncthreads();
    s[tid] += t;
    __syncthreads();
  }
  float base = (tid > 0) ? s[tid - 1] : 0.f;
#pragma unroll
  for (int i = 0; i < 8; ++i) {
    int t = tbase + i;
    int m = b * T_SEQ + t;
    float r = a[i] + base;
    R[(size_t)m * DHEAD + h] = r;
    if (layer == NLAYERS - 1)
      Rbf[(size_t)m * DHEAD + h] = (__bf16)(__expf(q[(size_t)m * 4 + 3]) * r);
  }
}

// ---------------------------------------------------------------------------
// K4: Xo[m][n] (bf16) = sum_k Rbf[m][k] * WOt[n][k].  K=512.  (1.6% of FLOPs.)
// ---------------------------------------------------------------------------
__global__ void gemm_ro_kernel(const __bf16* __restrict__ Rb,
                               const __bf16* __restrict__ WOt,
                               __bf16* __restrict__ Xo) {
  int lane = threadIdx.x & 31, wave = threadIdx.x >> 5;
  int mtile = blockIdx.x * 32 + (wave >> 2) * 16;
  int ntile = blockIdx.y * 64 + (wave & 3) * 16;
  int lm = lane & 15;
  const __bf16* pa = Rb  + (size_t)(mtile + lm) * DHEAD + ((lane >= 16) ? 8 : 0);
  const __bf16* pb = WOt + (size_t)(ntile + lm) * DHEAD + ((lane >= 16) ? 8 : 0);

  f32x8 c;
#pragma unroll
  for (int i = 0; i < 8; ++i) c[i] = 0.f;

  for (int kt = 0; kt < DHEAD; kt += 32) {
    bf16x16 a = load_a_at(pa, kt);
    bf16x16 b = load_b_at(pb, kt);
    c = __builtin_amdgcn_wmma_f32_16x16x32_bf16(false, a, false, b, (short)0, c,
                                                false, false);
  }
  int mo = (lane >= 16) ? 8 : 0;
#pragma unroll
  for (int r = 0; r < 8; ++r)
    Xo[(size_t)(mtile + r + mo) * DHID + ntile + lm] = (__bf16)c[r];
}

// ---------------------------------------------------------------------------
// K5: unembedding (dominant, 268 GFLOP).  4x4 register blocking per wave
// (32 FLOP/byte of cache traffic), hoisted per-lane base pointers so k-tile
// offsets fold into load immediates, launch_bounds(256,1) to use the large
// VGPR file without spilling.  Fully-unrolled epilogue (static accumulator
// indexing -> no scratch demotion): bias + LDS transpose so the 524 MB fp32
// output is written t-contiguous (coalesced).
// ---------------------------------------------------------------------------
__global__ __launch_bounds__(256, 1)
void gemm_wu_kernel(const __bf16* __restrict__ Xo,
                    const __bf16* __restrict__ WUt,
                    const float* __restrict__ bU,
                    float* __restrict__ out) {
  __shared__ float tile[8][16][17];
  int lane = threadIdx.x & 31, wave = threadIdx.x >> 5;
  int mtile = blockIdx.x * 128 + (wave >> 2) * 64;
  int ntile = blockIdx.y * 256 + (wave & 3) * 64;
  int lm  = lane & 15;
  int hiA = (lane >= 16) ? 8 : 0;
  int hiB = (lane >= 16) ? 8 : 0;

  float bias[4];
#pragma unroll
  for (int j = 0; j < 4; ++j) bias[j] = bU[ntile + j * 16 + lm];

  const __bf16* pa[4];
  const __bf16* pb[4];
#pragma unroll
  for (int i = 0; i < 4; ++i)
    pa[i] = Xo + (size_t)(mtile + i * 16 + lm) * DHID + hiA;
#pragma unroll
  for (int j = 0; j < 4; ++j)
    pb[j] = WUt + (size_t)(ntile + j * 16 + lm) * DHID + hiB;

  f32x8 c[4][4];
#pragma unroll
  for (int i = 0; i < 4; ++i)
#pragma unroll
    for (int j = 0; j < 4; ++j)
#pragma unroll
      for (int r = 0; r < 8; ++r) c[i][j][r] = 0.f;

#pragma unroll 1
  for (int kt = 0; kt < DHID; kt += 32) {
    bf16x16 a[4], b[4];
#pragma unroll
    for (int i = 0; i < 4; ++i) a[i] = load_a_at(pa[i], kt);
#pragma unroll
    for (int j = 0; j < 4; ++j) b[j] = load_b_at(pb[j], kt);
#pragma unroll
    for (int i = 0; i < 4; ++i)
#pragma unroll
      for (int j = 0; j < 4; ++j)
        c[i][j] = __builtin_amdgcn_wmma_f32_16x16x32_bf16(
            false, a[i], false, b[j], (short)0, c[i][j], false, false);
  }

  int mo = (lane >= 16) ? 8 : 0;
  int nl = lane >> 1;
  int mh = (lane & 1) * 8;
#pragma unroll
  for (int i = 0; i < 4; ++i) {
#pragma unroll
    for (int j = 0; j < 4; ++j) {
      __syncthreads();
#pragma unroll
      for (int r = 0; r < 8; ++r)
        tile[wave][r + mo][lm] = c[i][j][r] + bias[j];
      __syncthreads();
      int n  = ntile + j * 16 + nl;
      int m0 = mtile + i * 16 + mh;
      int b  = m0 >> 11;             // /T_SEQ
      int t  = m0 & (T_SEQ - 1);
      float* dst = out + (size_t)b * VOUT * T_SEQ + (size_t)n * T_SEQ + t;
#pragma unroll
      for (int u = 0; u < 8; ++u) dst[u] = tile[wave][mh + u][nl];
    }
  }
}

// ---------------------------------------------------------------------------
// Launcher
// ---------------------------------------------------------------------------
extern "C" void kernel_launch(void* const* d_in, const int* in_sizes, int n_in,
                              void* d_out, int out_size, void* d_ws, size_t ws_size,
                              hipStream_t stream) {
  (void)in_sizes; (void)n_in; (void)out_size; (void)ws_size;
  const int*   tokens = (const int*)  d_in[0];
  const float* emb    = (const float*)d_in[1];
  const float* pos    = (const float*)d_in[2];
  const float* WQ     = (const float*)d_in[3];
  const float* WK     = (const float*)d_in[4];
  const float* WV     = (const float*)d_in[5];
  const float* WO     = (const float*)d_in[6];
  const float* WU     = (const float*)d_in[7];
  const float* bU     = (const float*)d_in[8];
  float* out = (float*)d_out;

  // Workspace layout (all offsets 256-aligned).
  char* ws = (char*)d_ws;
  size_t o = 0;
  __bf16* xbf = (__bf16*)(ws + o); o += (size_t)MTOT * DHID * 2;          // 8 MB
  float*  qv  = (float*) (ws + o); o += (size_t)MTOT * 4 * 4;             // 64 KB
  float*  kv  = (float*) (ws + o); o += (size_t)MTOT * 4 * 4;             // 64 KB
  __bf16* WVt = (__bf16*)(ws + o); o += (size_t)NV * DHID * 2;            // 4 MB
  float*  Vt  = (float*) (ws + o); o += (size_t)NV * MTOT * 4;            // 32 MB
  float*  R   = (float*) (ws + o); o += (size_t)MTOT * DHEAD * 4;         // 8 MB
  __bf16* Rbf = (__bf16*)(ws + o); o += (size_t)MTOT * DHEAD * 2;         // 4 MB
  __bf16* WOt = (__bf16*)(ws + o); o += (size_t)DHID * DHEAD * 2;         // 1 MB
  __bf16* Xo  = (__bf16*)(ws + o); o += (size_t)MTOT * DHID * 2;          // 8 MB
  __bf16* WUt = (__bf16*)(ws + o); o += (size_t)VOUT * DHID * 2;          // 62.5 MB

  // K0: embeddings + q,k scalars
  embed_qk_kernel<<<MTOT, 256, 0, stream>>>(tokens, emb, pos, WQ, WK, xbf, qv, kv);

  // K1: weight transposes to bf16 N-major (K-contiguous for B fragments)
  transpose_f32_bf16<<<dim3(DHEAD / 32, DHID / 32, NLAYERS), dim3(32, 8), 0, stream>>>(
      WV, WVt, DHID, DHEAD, (size_t)DHID * DHEAD, (size_t)DHEAD * DHID);
  transpose_f32_bf16<<<dim3(DHID / 32, DHEAD / 32, 1), dim3(32, 8), 0, stream>>>(
      WO, WOt, DHEAD, DHID, 0, 0);
  transpose_f32_bf16<<<dim3(VOUT / 32, DHID / 32, 1), dim3(32, 8), 0, stream>>>(
      WU, WUt, DHID, VOUT, 0, 0);

  // K2: V projection (WMMA, 4x4 blocked), transposed store
  gemm_xv_kernel<<<dim3(MTOT / 128, NV / 256), 256, 0, stream>>>(xbf, WVt, Vt);

  // K3: LISS scan, 4 layer passes
  for (int l = 0; l < NLAYERS; ++l)
    scan_layer_kernel<<<BATCH * DHEAD, 256, 0, stream>>>(Vt, qv, kv, R, Rbf, l);

  // K4: W_O projection (WMMA) -> bf16 activations
  gemm_ro_kernel<<<dim3(MTOT / 32, DHID / 64), 256, 0, stream>>>(Rbf, WOt, Xo);

  // K5: unembedding GEMM (WMMA, 4x4 blocked) + bias + coalesced store
  gemm_wu_kernel<<<dim3(MTOT / 128, VOUT / 256), 256, 0, stream>>>(Xo, WUt, bU, out);
}
